// SparseCoder_76897094467600
// MI455X (gfx1250) — matrix operand order repaired and checked by reading, not compiled, not run on hardware
//
#include <hip/hip_runtime.h>
#include <hip/hip_bf16.h>
#include <stdint.h>

// Problem dimensions (fixed by the reference setup)
#define NROWS 8192
#define DIM   2048
#define NLAT  32768
#define TOPK  32
#define ENC_TILE 256   // latents per workgroup tile (8 waves x 32)

typedef __attribute__((ext_vector_type(16))) __bf16 bf16x16;
typedef __attribute__((ext_vector_type(8)))  float  f32x8;

// ---------------------------------------------------------------------------
// fp32 -> bf16 (round to nearest even), vectorized x4
// ---------------------------------------------------------------------------
__device__ __forceinline__ unsigned short f2bf(float f) {
    unsigned int u = __float_as_uint(f);
    u += 0x7FFFu + ((u >> 16) & 1u);
    return (unsigned short)(u >> 16);
}

__global__ __launch_bounds__(256) void cvt_f32_bf16_kernel(
    const float4* __restrict__ src, ushort4* __restrict__ dst, long n4)
{
    long i = (long)blockIdx.x * blockDim.x + threadIdx.x;
    long stride = (long)gridDim.x * blockDim.x;
    for (; i < n4; i += stride) {
        float4 f = src[i];
        ushort4 o;
        o.x = f2bf(f.x); o.y = f2bf(f.y); o.z = f2bf(f.z); o.w = f2bf(f.w);
        dst[i] = o;
    }
}

// ---------------------------------------------------------------------------
// Wave helpers (wave32)
// ---------------------------------------------------------------------------
__device__ __forceinline__ float wred_min(float v) {
    for (int m = 16; m; m >>= 1) v = fminf(v, __shfl_xor(v, m, 32));
    return v;
}
__device__ __forceinline__ float wred_max(float v) {
    for (int m = 16; m; m >>= 1) v = fmaxf(v, __shfl_xor(v, m, 32));
    return v;
}

// Bitonic sort of 32 (value,index) pairs across a wave: value desc, index asc
__device__ __forceinline__ void sort32_desc(float& v, int& idx) {
    const int lane = threadIdx.x & 31;
    for (int k = 2; k <= 32; k <<= 1) {
        for (int j = k >> 1; j >= 1; j >>= 1) {
            float ov = __shfl_xor(v, j, 32);
            int   oi = __shfl_xor(idx, j, 32);
            bool iGreater = (v > ov) || (v == ov && idx < oi);
            bool isLow = ((lane & j) == 0);
            bool descBlock = ((lane & k) == 0);
            bool keep = (isLow == descBlock) ? iGreater : !iGreater;
            if (!keep) { v = ov; idx = oi; }
        }
    }
}

// ---------------------------------------------------------------------------
// bf16 WMMA VGPR layouts used here (contiguous k, same for A and B; ISA table order differs):
//   A (16-bit 16x32): lane m=l&15, half=l>>4; v0..3 hold K=half*16+0..7 packed,
//                     v4..7 hold K=half*16+8..15  -> two b128 loads per lane.
//   B (16-bit 32x16): lane n=l&15, koff=16*half; v holds K=koff+2v,2v+1
//                     -> 16 contiguous K elements -> two b128 loads per lane.
// ---------------------------------------------------------------------------

// 16x32 tile (used by the skip GEMM)
__device__ __forceinline__ void wmma_tile_16x32(
    const unsigned short* __restrict__ aRow,
    const unsigned short* __restrict__ b0Row,
    const unsigned short* __restrict__ b1Row,
    int half, f32x8& acc0, f32x8& acc1)
{
#pragma unroll 4
    for (int k = 0; k < DIM; k += 32) {
        union { uint4 u[2]; bf16x16 v; } a, b0, b1;
        const unsigned short* ap = aRow + k + half * 16;
        a.u[0] = *(const uint4*)(ap);
        a.u[1] = *(const uint4*)(ap + 8);
        const unsigned short* p0 = b0Row + k + half * 16;
        b0.u[0] = *(const uint4*)(p0);
        b0.u[1] = *(const uint4*)(p0 + 8);
        const unsigned short* p1 = b1Row + k + half * 16;
        b1.u[0] = *(const uint4*)(p1);
        b1.u[1] = *(const uint4*)(p1 + 8);
        acc0 = __builtin_amdgcn_wmma_f32_16x16x32_bf16(
                   false, a.v, false, b0.v, (short)0, acc0, false, false);
        acc1 = __builtin_amdgcn_wmma_f32_16x16x32_bf16(
                   false, a.v, false, b1.v, (short)0, acc1, false, false);
    }
}

// 32x32 output quad: 2 A row-tiles (LDS, interleaved so both sit within the
// 16-bit DS immediate-offset range of one base pointer) x 2 B col-tiles
// (global), 4 accumulators.  1 global b128 + 1 ds_load_b128 per WMMA.
__device__ __forceinline__ void wmma_quad_32x32(
    const unsigned short* aPair,               // LDS: rows (m16, m16+16) pair
    const unsigned short* __restrict__ b0Row,  // global: latent jb + n16
    const unsigned short* __restrict__ b1Row,  // global: latent jb+16 + n16
    int half, f32x8& acc00, f32x8& acc01, f32x8& acc10, f32x8& acc11)
{
#pragma unroll 2
    for (int k = 0; k < DIM; k += 32) {
        union { uint4 u[2]; bf16x16 v; } a0, a1, b0, b1;
        const unsigned short* ap0 = aPair + k + half * 16;
        a0.u[0] = *(const uint4*)(ap0);
        a0.u[1] = *(const uint4*)(ap0 + 8);
        const unsigned short* ap1 = ap0 + DIM;           // +4096 B: DS offset
        a1.u[0] = *(const uint4*)(ap1);
        a1.u[1] = *(const uint4*)(ap1 + 8);
        const unsigned short* p0 = b0Row + k + half * 16;
        b0.u[0] = *(const uint4*)(p0);
        b0.u[1] = *(const uint4*)(p0 + 8);
        const unsigned short* p1 = b1Row + k + half * 16;
        b1.u[0] = *(const uint4*)(p1);
        b1.u[1] = *(const uint4*)(p1 + 8);
        acc00 = __builtin_amdgcn_wmma_f32_16x16x32_bf16(
                    false, a0.v, false, b0.v, (short)0, acc00, false, false);
        acc01 = __builtin_amdgcn_wmma_f32_16x16x32_bf16(
                    false, a0.v, false, b1.v, (short)0, acc01, false, false);
        acc10 = __builtin_amdgcn_wmma_f32_16x16x32_bf16(
                    false, a1.v, false, b0.v, (short)0, acc10, false, false);
        acc11 = __builtin_amdgcn_wmma_f32_16x16x32_bf16(
                    false, a1.v, false, b1.v, (short)0, acc11, false, false);
    }
}

// ---------------------------------------------------------------------------
// Encoder GEMM (bf16 WMMA, fp32 acc) fused with bias+relu+top-32.
// Block = 256 threads (8 waves) owns 32 rows; sweeps all 32768 latents in
// tiles of 256 (each wave: 32 rows x 32 latents = 4 WMMA accumulators).
// The 32x2048 bf16 x-tile (128 KB) is staged into LDS once with rows m and
// m+16 interleaved (lrow = 2*(m&15) + (m>>4)) so each lane's two A tiles are
// base and base+4096B -> all DS accesses use immediate offsets.
// ---------------------------------------------------------------------------
__global__ __launch_bounds__(256) void encode_topk_kernel(
    const unsigned short* __restrict__ xbf,
    const unsigned short* __restrict__ wencbf,
    const float* __restrict__ b_enc,
    float* __restrict__ out_acts,
    int*   __restrict__ out_idx)
{
    __shared__ unsigned short tileA[32 * DIM];   // 128 KB (of 320 KB/WGP)
    __shared__ float cand[32][ENC_TILE];         // 32 KB
    __shared__ float topv[32][TOPK];             // 4 KB
    __shared__ int   topi[32][TOPK];             // 4 KB

    const int tid  = threadIdx.x;
    const int lane = tid & 31;
    const int wave = tid >> 5;       // 0..7: latent sub-block
    const int rowbase = blockIdx.x * 32;

    // stage x-tile with row interleave: global row m -> LDS row 2*(m&15)+(m>>4)
    {
        const int segs = DIM / 8;    // uint4 segments per row
        const uint4* __restrict__ src = (const uint4*)(xbf + (long)rowbase * DIM);
        uint4* dst = (uint4*)tileA;
        for (int i = tid; i < 32 * segs; i += 256) {
            const int row = i / segs, seg = i - row * segs;
            const int lrow = ((row & 15) << 1) | (row >> 4);
            dst[lrow * segs + seg] = src[i];
        }
    }

    for (int i = tid; i < 32 * TOPK; i += 256) {
        (&topv[0][0])[i] = -3.4e38f;
        (&topi[0][0])[i] = 0;
    }
    __syncthreads();

    const int half = lane >> 4;
    const int n16  = lane & 15;
    const unsigned short* aPair = tileA + (n16 * 2) * DIM;  // rows n16, n16+16

    for (int t = 0; t < NLAT / ENC_TILE; ++t) {
        const int jb = t * ENC_TILE + wave * 32;
        f32x8 acc00 = {}, acc01 = {}, acc10 = {}, acc11 = {};
        const unsigned short* b0Row = wencbf + (long)(jb + n16) * DIM;
        const unsigned short* b1Row = wencbf + (long)(jb + 16 + n16) * DIM;
        wmma_quad_32x32(aPair, b0Row, b1Row, half,
                        acc00, acc01, acc10, acc11);

        // epilogue: bias + relu -> LDS candidate tile (32 rows x 256 latents)
        const float be0 = b_enc[jb + n16];
        const float be1 = b_enc[jb + 16 + n16];
        const int rl = 8 * half;
        const int c0 = wave * 32 + n16;
#pragma unroll
        for (int c = 0; c < 8; ++c) {
            float v00 = acc00[c] + be0; v00 = v00 > 0.f ? v00 : 0.f;
            float v01 = acc01[c] + be1; v01 = v01 > 0.f ? v01 : 0.f;
            float v10 = acc10[c] + be0; v10 = v10 > 0.f ? v10 : 0.f;
            float v11 = acc11[c] + be1; v11 = v11 > 0.f ? v11 : 0.f;
            cand[rl + c][c0]           = v00;
            cand[rl + c][c0 + 16]      = v01;
            cand[rl + c + 16][c0]      = v10;
            cand[rl + c + 16][c0 + 16] = v11;
        }
        __syncthreads();

        // merge: wave handles 4 rows; top-32 pairs live one-per-lane
        const int lbase = t * ENC_TILE;
        for (int rr = 0; rr < 4; ++rr) {
            const int r = wave * 4 + rr;
            float tv = topv[r][lane];
            int   ti = topi[r][lane];
            float rmin = wred_min(tv);
            float lm = cand[r][lane];
#pragma unroll
            for (int i = 1; i < ENC_TILE / 32; ++i)
                lm = fmaxf(lm, cand[r][lane + 32 * i]);
            const float tmax = wred_max(lm);
            if (tmax > rmin) {                 // wave-uniform
                for (int c = 0; c < ENC_TILE; ++c) {
                    const float v = cand[r][c];    // uniform broadcast read
                    if (v > rmin) {                // wave-uniform
                        // argmin lane of current state
                        float mv = tv; int ml = lane;
                        for (int m = 16; m; m >>= 1) {
                            float ov = __shfl_xor(mv, m, 32);
                            int   ol = __shfl_xor(ml, m, 32);
                            if (ov < mv || (ov == mv && ol < ml)) { mv = ov; ml = ol; }
                        }
                        if (lane == ml) { tv = v; ti = lbase + c; }
                        rmin = wred_min(tv);
                    }
                }
                topv[r][lane] = tv;
                topi[r][lane] = ti;
            }
        }
        __syncthreads();
    }

    // sort each row descending (index tiebreak asc, matching lax.top_k) + emit
    for (int rr = 0; rr < 4; ++rr) {
        const int r = wave * 4 + rr;
        float tv = topv[r][lane];
        int   ti = topi[r][lane];
        sort32_desc(tv, ti);
        const int grow = rowbase + r;
        out_acts[grow * TOPK + lane] = tv;
        out_idx[grow * TOPK + lane]  = ti;
    }
}

// ---------------------------------------------------------------------------
// Skip GEMM: sae_out = x @ W_skip^T + b_dec   (bf16 WMMA, fp32 out)
// grid = (NROWS/32, DIM/128); waves: rb = w>>2 row block, cs = w&3 col block.
// ---------------------------------------------------------------------------
__global__ __launch_bounds__(256) void skip_gemm_kernel(
    const unsigned short* __restrict__ xbf,
    const unsigned short* __restrict__ wskipbf,
    const float* __restrict__ b_dec,
    float* __restrict__ out)
{
    const int tid  = threadIdx.x;
    const int lane = tid & 31;
    const int wave = tid >> 5;
    const int rb   = wave >> 2;
    const int cs   = wave & 3;
    const int rowbase = blockIdx.x * 32;
    const int colbase = blockIdx.y * 128;

    const int half = lane >> 4;
    const int n16  = lane & 15;
    const unsigned short* aRow  = xbf + (long)(rowbase + rb * 16 + n16) * DIM;
    const int jb = colbase + cs * 32;
    const unsigned short* b0Row = wskipbf + (long)(jb + n16) * DIM;
    const unsigned short* b1Row = wskipbf + (long)(jb + 16 + n16) * DIM;

    f32x8 acc0 = {}, acc1 = {};
    wmma_tile_16x32(aRow, b0Row, b1Row, half, acc0, acc1);

    const int col0 = jb + n16;
    const int col1 = col0 + 16;
    const float bd0 = b_dec[col0];
    const float bd1 = b_dec[col1];
#pragma unroll
    for (int c = 0; c < 8; ++c) {
        const int row = rowbase + rb * 16 + 8 * half + c;
        out[(long)row * DIM + col0] = acc0[c] + bd0;
        out[(long)row * DIM + col1] = acc1[c] + bd1;
    }
}

// ---------------------------------------------------------------------------
// Sparse decode: out[r,:] += sum_k acts[r,k] * W_dec[idx[r,k], :]
// One wave per row; acts/idx held one-per-lane and broadcast via shuffles.
// ---------------------------------------------------------------------------
__global__ __launch_bounds__(256) void sparse_decode_kernel(
    const float* __restrict__ acts,
    const int*   __restrict__ idx,
    const float* __restrict__ Wdec,
    float* __restrict__ out)
{
    const int lane = threadIdx.x & 31;
    const int wave = threadIdx.x >> 5;
    const int row  = blockIdx.x * 8 + wave;

    const float a  = acts[row * TOPK + lane];
    const int   id = idx[row * TOPK + lane];

    for (int chunk = 0; chunk < DIM; chunk += 256) {
        float accv[8];
        const long base = (long)row * DIM + chunk + lane;
#pragma unroll
        for (int t = 0; t < 8; ++t) accv[t] = out[base + 32 * t];
        for (int k = 0; k < TOPK; ++k) {
            const float ak = __shfl(a, k, 32);
            const int   ik = __shfl(id, k, 32);
            const float* wr = Wdec + (long)ik * DIM + chunk + lane;
#pragma unroll
            for (int t = 0; t < 8; ++t) accv[t] = fmaf(ak, wr[32 * t], accv[t]);
        }
#pragma unroll
        for (int t = 0; t < 8; ++t) out[base + 32 * t] = accv[t];
    }
}

// ---------------------------------------------------------------------------
// FVU: column sums of y, then per-block partial (l2, var), then fixed-order
// final reduce.  No atomics -> deterministic.
// ---------------------------------------------------------------------------
__global__ __launch_bounds__(256) void colsum_kernel(
    const float* __restrict__ y, float* __restrict__ colsum)
{
    const int j = blockIdx.x * blockDim.x + threadIdx.x;   // 2048 threads
    float s = 0.f;
    for (int n = 0; n < NROWS; ++n) s += y[(long)n * DIM + j];
    colsum[j] = s;
}

__global__ __launch_bounds__(256) void loss_partial_kernel(
    const float* __restrict__ y, const float* __restrict__ out,
    const float* __restrict__ colsum,
    float* __restrict__ p_l2, float* __restrict__ p_var)
{
    const long total = (long)NROWS * DIM;
    long i = (long)blockIdx.x * blockDim.x + threadIdx.x;
    const long stride = (long)gridDim.x * blockDim.x;
    float l2 = 0.f, var = 0.f;
    for (; i < total; i += stride) {
        const int j = (int)(i & (DIM - 1));
        const float yy = y[i];
        const float e = yy - out[i];
        l2 += e * e;
        const float d = yy - colsum[j] * (1.0f / NROWS);
        var += d * d;
    }
    for (int m = 16; m; m >>= 1) {
        l2  += __shfl_xor(l2, m, 32);
        var += __shfl_xor(var, m, 32);
    }
    __shared__ float sl[8], sv[8];
    const int lane = threadIdx.x & 31, wave = threadIdx.x >> 5;
    if (lane == 0) { sl[wave] = l2; sv[wave] = var; }
    __syncthreads();
    if (threadIdx.x == 0) {
        float tl = 0.f, tv = 0.f;
        for (int w = 0; w < 8; ++w) { tl += sl[w]; tv += sv[w]; }
        p_l2[blockIdx.x] = tl;
        p_var[blockIdx.x] = tv;
    }
}

__global__ void fvu_final_kernel(const float* __restrict__ p_l2,
                                 const float* __restrict__ p_var,
                                 int n, float* __restrict__ fvu)
{
    if (threadIdx.x == 0 && blockIdx.x == 0) {
        float l2 = 0.f, var = 0.f;
        for (int i = 0; i < n; ++i) { l2 += p_l2[i]; var += p_var[i]; }
        fvu[0] = l2 / var;
    }
}

// ---------------------------------------------------------------------------
// Launcher
// ---------------------------------------------------------------------------
extern "C" void kernel_launch(void* const* d_in, const int* in_sizes, int n_in,
                              void* d_out, int out_size, void* d_ws, size_t ws_size,
                              hipStream_t stream)
{
    const float* x      = (const float*)d_in[0];
    const float* y      = (const float*)d_in[1];
    const float* W_enc  = (const float*)d_in[2];
    const float* b_enc  = (const float*)d_in[3];
    const float* W_dec  = (const float*)d_in[4];
    const float* b_dec  = (const float*)d_in[5];
    const float* W_skip = (const float*)d_in[6];

    float* out         = (float*)d_out;
    float* sae_out     = out;                                     // N*D
    float* latent_acts = out + (size_t)NROWS * DIM;               // N*K
    int*   latent_idx  = (int*)(latent_acts + (size_t)NROWS * TOPK);
    float* fvu         = (float*)(latent_idx + (size_t)NROWS * TOPK);

    // workspace layout (bf16 copies + reduction scratch), all 256B aligned
    unsigned short* xbf     = (unsigned short*)d_ws;
    unsigned short* wencbf  = xbf    + (size_t)NROWS * DIM;       // 32 MB in
    unsigned short* wskipbf = wencbf + (size_t)NLAT  * DIM;       // +128 MB
    float* colsum = (float*)(wskipbf + (size_t)DIM * DIM);        // +8 MB
    float* p_l2   = colsum + DIM;
    float* p_var  = p_l2 + 2048;

    // 1) fp32 -> bf16 conversions (W_enc bf16 = 128 MB -> stays L2-resident)
    cvt_f32_bf16_kernel<<<2048, 256, 0, stream>>>(
        (const float4*)x, (ushort4*)xbf, (long)NROWS * DIM / 4);
    cvt_f32_bf16_kernel<<<2048, 256, 0, stream>>>(
        (const float4*)W_enc, (ushort4*)wencbf, (long)NLAT * DIM / 4);
    cvt_f32_bf16_kernel<<<2048, 256, 0, stream>>>(
        (const float4*)W_skip, (ushort4*)wskipbf, (long)DIM * DIM / 4);

    // 2) encoder GEMM + fused top-32 (the 1.1 TFLOP hot loop, bf16 WMMA)
    encode_topk_kernel<<<NROWS / 32, 256, 0, stream>>>(
        xbf, wencbf, b_enc, latent_acts, latent_idx);

    // 3) skip GEMM + b_dec -> sae_out
    skip_gemm_kernel<<<dim3(NROWS / 32, DIM / 128), 256, 0, stream>>>(
        xbf, wskipbf, b_dec, sae_out);

    // 4) sparse decode accumulate
    sparse_decode_kernel<<<NROWS / 8, 256, 0, stream>>>(
        latent_acts, latent_idx, W_dec, sae_out);

    // 5) FVU
    colsum_kernel<<<DIM / 256, 256, 0, stream>>>(y, colsum);
    loss_partial_kernel<<<2048, 256, 0, stream>>>(y, sae_out, colsum, p_l2, p_var);
    fvu_final_kernel<<<1, 32, 0, stream>>>(p_l2, p_var, 2048, fvu);
}
